// PartitionAttentionCl_16827681866190
// MI455X (gfx1250) — matrix operand order repaired and checked by reading, not compiled, not run on hardware
//
#include <hip/hip_runtime.h>
#include <hip/hip_bf16.h>
#include <math.h>

typedef __bf16 bf16_t;
typedef __attribute__((ext_vector_type(16))) __bf16 v16bf;
typedef __attribute__((ext_vector_type(8)))  __bf16 v8bf;
typedef __attribute__((ext_vector_type(8)))  float  v8f;
typedef __attribute__((ext_vector_type(4)))  int    v4i;

union Frag16 { v16bf v; v8bf h[2]; };

#define QSCALE 0.17677669529663687f  // 1/sqrt(32)

// ---------------------------------------------------------------------------
// gfx1250 async global->LDS copy (ASYNCcnt-tracked DMA, no VGPR round trip).
// Probe (round 2 diagnostic) showed the builtin's param 0 is an AS(1) int4*.
// AS(3) pointer is formed by 32-bit truncation of the flat address (ISA 10.2:
// flat->LDS mapping discards the upper bits).
// ---------------------------------------------------------------------------
#if defined(__gfx1250__) && __has_builtin(__builtin_amdgcn_global_load_async_to_lds_b128)
#define HAVE_ASYNC_LDS 1
#else
#define HAVE_ASYNC_LDS 0
#endif

#if HAVE_ASYNC_LDS
__device__ __forceinline__ void cp_async16(void* lds, const void* g) {
  __builtin_amdgcn_global_load_async_to_lds_b128(
      (__attribute__((address_space(1))) v4i*)(unsigned long long)g,
      (__attribute__((address_space(3))) v4i*)(unsigned int)(unsigned long long)lds,
      0, 0);
}
#if __has_builtin(__builtin_amdgcn_s_wait_asynccnt)
#define ASYNC_WAIT() __builtin_amdgcn_s_wait_asynccnt(0)
#else
#define ASYNC_WAIT() asm volatile("s_wait_asynccnt 0" ::: "memory")
#endif
#else
__device__ __forceinline__ void cp_async16(void* lds, const void* g) {
  *(v8bf*)lds = *(const v8bf*)g;   // sync fallback; __syncthreads covers DScnt
}
#define ASYNC_WAIT() do {} while (0)
#endif

// ---------------------------------------------------------------------------
// Small prep kernels
// ---------------------------------------------------------------------------
__global__ __launch_bounds__(256) void cvt_bf16_kernel(const float* __restrict__ src,
                                                       bf16_t* __restrict__ dst, int n) {
  int i = blockIdx.x * 256 + threadIdx.x;
  if (i < n) dst[i] = (bf16_t)src[i];
}

// biasE[h][n][m] = table[rel_idx(n,m)][h],  rel_idx = (yi-yj+7)*15 + (xi-xj+7)
__global__ __launch_bounds__(256) void bias_expand_kernel(const float* __restrict__ table,
                                                          float* __restrict__ out) {
  int i = blockIdx.x * 256 + threadIdx.x;   // 8*64*64 = 32768
  int h = i >> 12, n = (i >> 6) & 63, m = i & 63;
  int yi = n >> 3, xi = n & 7, yj = m >> 3, xj = m & 7;
  int idx = (yi - yj + 7) * 15 + (xi - xj + 7);
  out[i] = table[idx * 8 + h];
}

// ---------------------------------------------------------------------------
// LayerNorm over C=256, one wave per token (8 tokens / block).
// WINDOW=true permutes tokens into window-major order for the attention path.
// ---------------------------------------------------------------------------
template <bool WINDOW>
__global__ __launch_bounds__(256)
void ln_kernel(const float* __restrict__ X, const float* __restrict__ w,
               const float* __restrict__ b, bf16_t* __restrict__ Y) {
  const int lane = threadIdx.x & 31;
  const int t = blockIdx.x * 8 + (threadIdx.x >> 5);
  const float* xp = X + (size_t)t * 256;
  float v[8], s = 0.f, ss = 0.f;
#pragma unroll
  for (int u = 0; u < 8; ++u) {
    v[u] = xp[lane + 32 * u];
    s += v[u]; ss += v[u] * v[u];
  }
#pragma unroll
  for (int o = 16; o; o >>= 1) { s += __shfl_xor(s, o, 32); ss += __shfl_xor(ss, o, 32); }
  const float mean = s * (1.f / 256.f);
  const float var  = ss * (1.f / 256.f) - mean * mean;
  const float rstd = rsqrtf(var + 1e-6f);

  size_t dst;
  if (WINDOW) {
    int bb = t >> 12, ii = (t >> 6) & 63, jj = t & 63;
    int win = bb * 64 + (ii >> 3) * 8 + (jj >> 3);
    int tok = (ii & 7) * 8 + (jj & 7);
    dst = ((size_t)win * 64 + tok) * 256;
  } else {
    dst = (size_t)t * 256;
  }
#pragma unroll
  for (int u = 0; u < 8; ++u) {
    int c = lane + 32 * u;
    Y[dst + c] = (bf16_t)((v[u] - mean) * rstd * w[c] + b[c]);
  }
}

// ---------------------------------------------------------------------------
// Generic bf16 WMMA GEMM:  out[m][n] = sum_k A[m][k] * Bw[n][k] + bias[n]
// Block tile 64x256, K step 32, double-buffered LDS with async global->LDS
// staging. 8 waves, each owns a 32x64 patch (2x4 tiles, 8 WMMA per K step).
// MODE 0: QKV scatter (q scaled, v transposed)   MODE 1: proj + residual (unwindow)
// MODE 2: GELU -> bf16                           MODE 3: + residual -> fp32 d_out
// ---------------------------------------------------------------------------
template <int MODE>
__global__ __launch_bounds__(256)
void gemm_bf16_kernel(const bf16_t* __restrict__ A, const bf16_t* __restrict__ Bw,
                      const float* __restrict__ bias, const float* __restrict__ aux,
                      void* __restrict__ out0, void* __restrict__ out1,
                      void* __restrict__ out2, int M, int N, int K) {
  __shared__ bf16_t As[2][64][40];    // padded rows: 80 B (16B-aligned chunks)
  __shared__ bf16_t Bs[2][256][40];

  const int tid  = threadIdx.x;
  const int lane = tid & 31;
  const int wv   = tid >> 5;
  const int l15  = lane & 15;
  const int hi   = lane >> 4;
  const int m0   = blockIdx.x * 64;
  const int n0   = blockIdx.y * 256;
  const int wm   = (wv & 1) * 32;   // wave's 32-row patch
  const int wn   = (wv >> 1) * 64;  // wave's 64-col patch

  const int arow = tid >> 2, acol = (tid & 3) * 8;  // A tile: 64 x 32
  const int brow = tid;                             // B tile: 256 x 32 (4x16B/thread)

  // stage one 64x32 A tile + one 256x32 B tile into LDS buffer `buf`
  auto stage = [&](int buf, int k0) {
    cp_async16(&As[buf][arow][acol], A + (size_t)(m0 + arow) * K + k0 + acol);
    const bf16_t* bp = Bw + (size_t)(n0 + brow) * K + k0;
#pragma unroll
    for (int c = 0; c < 4; ++c)
      cp_async16(&Bs[buf][brow][c * 8], bp + c * 8);
  };

  v8f acc[2][4] = {};

  stage(0, 0);
  ASYNC_WAIT();
  __syncthreads();

  int p = 0;
  for (int k0 = 0; k0 < K; k0 += 32) {
    const bool more = (k0 + 32) < K;
    if (more) stage(p ^ 1, k0 + 32);   // DMA next tile while we compute

    Frag16 af[2], bfr[4];
#pragma unroll
    for (int i = 0; i < 2; ++i) {
      af[i].h[0] = *(const v8bf*)&As[p][wm + i * 16 + l15][hi * 8];
      af[i].h[1] = *(const v8bf*)&As[p][wm + i * 16 + l15][16 + hi * 8];
    }
#pragma unroll
    for (int j = 0; j < 4; ++j) {
      bfr[j].h[0] = *(const v8bf*)&Bs[p][wn + j * 16 + l15][hi * 8];
      bfr[j].h[1] = *(const v8bf*)&Bs[p][wn + j * 16 + l15][16 + hi * 8];
    }
#pragma unroll
    for (int i = 0; i < 2; ++i)
#pragma unroll
      for (int j = 0; j < 4; ++j)
        acc[i][j] = __builtin_amdgcn_wmma_f32_16x16x32_bf16(
            false, af[i].v, false, bfr[j].v, (short)0, acc[i][j], false, false);

    if (more) {
      ASYNC_WAIT();      // my async ops (next tile) have landed
      __syncthreads();   // everyone's have landed; old tile fully consumed
      p ^= 1;
    }
  }

#pragma unroll
  for (int i = 0; i < 2; ++i) {
#pragma unroll
    for (int j = 0; j < 4; ++j) {
#pragma unroll
      for (int r = 0; r < 8; ++r) {
        const int m = m0 + wm + i * 16 + hi * 8 + r;
        const int n = n0 + wn + j * 16 + l15;
        float val = acc[i][j][r] + bias[n];
        if (MODE == 0) {  // QKV scatter: n -> (head, which, d)
          int hh = n / 96, rem = n - hh * 96;
          int which = rem >> 5, d = rem & 31;
          int w = m >> 6, tok = m & 63;
          size_t qk = ((size_t)(w * 8 + hh) * 64 + tok) * 32 + d;    // [w][h][tok][d]
          if (which == 0)       ((bf16_t*)out0)[qk] = (bf16_t)(val * QSCALE);
          else if (which == 1)  ((bf16_t*)out1)[qk] = (bf16_t)val;
          else ((bf16_t*)out2)[((size_t)(w * 8 + hh) * 32 + d) * 64 + tok] = (bf16_t)val;
        } else if (MODE == 1) {  // proj + residual, window-major -> natural layout
          int w = m >> 6, t = m & 63;
          int bb = w >> 6, wh = (w >> 3) & 7, ww = w & 7;
          int ii = wh * 8 + (t >> 3), jj = ww * 8 + (t & 7);
          size_t nat = (((size_t)bb * 64 + ii) * 64 + jj) * 256 + n;
          ((float*)out0)[nat] = val + aux[nat];
        } else if (MODE == 2) {  // exact GELU -> bf16
          float g = 0.5f * val * (1.0f + erff(val * 0.70710678118654752f));
          ((bf16_t*)out0)[(size_t)m * N + n] = (bf16_t)g;
        } else {                 // MODE 3: + residual -> fp32 out
          size_t idx = (size_t)m * N + n;
          ((float*)out0)[idx] = val + aux[idx];
        }
      }
    }
  }
}

// ---------------------------------------------------------------------------
// Windowed attention: one block per (window, head).
// S = Q K^T + bias (16 WMMA tiles across 8 waves), softmax, O = P V (8 tiles).
// ---------------------------------------------------------------------------
__global__ __launch_bounds__(256)
void attn_kernel(const bf16_t* __restrict__ Q, const bf16_t* __restrict__ Km,
                 const bf16_t* __restrict__ Vm, const float* __restrict__ biasE,
                 bf16_t* __restrict__ O) {
  __shared__ float  S[64][68];
  __shared__ bf16_t P[64][72];   // 144 B rows: 16B-aligned fragment chunks

  const int tid = threadIdx.x, lane = tid & 31, wv = tid >> 5;
  const int l15 = lane & 15, hi = lane >> 4;
  const int wh = blockIdx.x;           // 0..8191
  const int w = wh >> 3, h = wh & 7;

  const bf16_t* q = Q + (size_t)wh * 64 * 32;   // [tok][d]
  const bf16_t* k = Km + (size_t)wh * 64 * 32;  // [tok][d]
  const bf16_t* v = Vm + (size_t)wh * 32 * 64;  // [d][tok]
  const float* bh = biasE + (size_t)h * 64 * 64;

  // ---- S = q @ k^T + bias (q pre-scaled by 1/sqrt(dh)) ----
#pragma unroll
  for (int tt = 0; tt < 2; ++tt) {
    const int t = wv * 2 + tt, tm = t >> 2, tn = t & 3;
    Frag16 a, bfr;
    const int qrow = tm * 16 + l15;
    a.h[0] = *(const v8bf*)(q + qrow * 32 + hi * 8);
    a.h[1] = *(const v8bf*)(q + qrow * 32 + 16 + hi * 8);
    const int kcol = tn * 16 + l15;
    bfr.h[0] = *(const v8bf*)(k + kcol * 32 + hi * 8);
    bfr.h[1] = *(const v8bf*)(k + kcol * 32 + 16 + hi * 8);
    v8f acc = {};
    acc = __builtin_amdgcn_wmma_f32_16x16x32_bf16(false, a.v, false, bfr.v,
                                                  (short)0, acc, false, false);
#pragma unroll
    for (int r = 0; r < 8; ++r) {
      const int row = tm * 16 + hi * 8 + r, col = tn * 16 + l15;
      S[row][col] = acc[r] + bh[row * 64 + col];
    }
  }
  __syncthreads();

  // ---- row softmax (each wave: 8 rows; each lane: cols lane, lane+32) ----
#pragma unroll
  for (int rr = 0; rr < 8; ++rr) {
    const int row = wv * 8 + rr;
    float a0 = S[row][lane], a1 = S[row][lane + 32];
    float mx = fmaxf(a0, a1);
#pragma unroll
    for (int o = 16; o; o >>= 1) mx = fmaxf(mx, __shfl_xor(mx, o, 32));
    float e0 = __expf(a0 - mx), e1 = __expf(a1 - mx);
    float sm = e0 + e1;
#pragma unroll
    for (int o = 16; o; o >>= 1) sm += __shfl_xor(sm, o, 32);
    const float inv = 1.0f / sm;
    P[row][lane]      = (bf16_t)(e0 * inv);
    P[row][lane + 32] = (bf16_t)(e1 * inv);
  }
  __syncthreads();

  // ---- O = P @ V  (M=64, N=32, K=64; one 16x16 tile per wave) ----
  const int tm = wv >> 1, tn = wv & 1;
  v8f acc = {};
#pragma unroll
  for (int kk = 0; kk < 64; kk += 32) {
    Frag16 a, bfr;
    const int prow = tm * 16 + l15;
    a.h[0] = *(const v8bf*)&P[prow][kk + hi * 8];
    a.h[1] = *(const v8bf*)&P[prow][kk + 16 + hi * 8];
    const int drow = tn * 16 + l15;
    bfr.h[0] = *(const v8bf*)(v + drow * 64 + kk + hi * 8);
    bfr.h[1] = *(const v8bf*)(v + drow * 64 + kk + 16 + hi * 8);
    acc = __builtin_amdgcn_wmma_f32_16x16x32_bf16(false, a.v, false, bfr.v,
                                                  (short)0, acc, false, false);
  }
#pragma unroll
  for (int r = 0; r < 8; ++r) {
    const int tok = tm * 16 + hi * 8 + r;
    const int c = h * 32 + tn * 16 + l15;
    O[((size_t)w * 64 + tok) * 256 + c] = (bf16_t)acc[r];  // proj A layout
  }
}

// ---------------------------------------------------------------------------
// Host-side launch
// ---------------------------------------------------------------------------
extern "C" void kernel_launch(void* const* d_in, const int* in_sizes, int n_in,
                              void* d_out, int out_size, void* d_ws, size_t ws_size,
                              hipStream_t stream) {
  (void)in_sizes; (void)n_in; (void)out_size; (void)ws_size;
  const float* x      = (const float*)d_in[0];
  const float* n1w    = (const float*)d_in[1];
  const float* n1b    = (const float*)d_in[2];
  const float* qkv_w  = (const float*)d_in[3];
  const float* qkv_b  = (const float*)d_in[4];
  const float* relTab = (const float*)d_in[5];
  const float* proj_w = (const float*)d_in[6];
  const float* proj_b = (const float*)d_in[7];
  const float* n2w    = (const float*)d_in[8];
  const float* n2b    = (const float*)d_in[9];
  const float* fc1_w  = (const float*)d_in[10];
  const float* fc1_b  = (const float*)d_in[11];
  const float* fc2_w  = (const float*)d_in[12];
  const float* fc2_b  = (const float*)d_in[13];
  float* out = (float*)d_out;

  const int M = 16 * 64 * 64;  // 65536 tokens
  char* ws = (char*)d_ws;

  // scratch layout (bytes, all 256-aligned)
  bf16_t* qkvW  = (bf16_t*)(ws + 0);               // 768*256*2   = 393216
  bf16_t* projW = (bf16_t*)(ws + 393216);          // 256*256*2   = 131072
  bf16_t* fc1W  = (bf16_t*)(ws + 524288);          // 1024*256*2  = 524288
  bf16_t* fc2W  = (bf16_t*)(ws + 1048576);         // 256*1024*2  = 524288
  float*  biasE = (float*) (ws + 1572864);         // 8*64*64*4   = 131072
  bf16_t* ywin  = (bf16_t*)(ws + 1703936);         // 65536*256*2 = 33554432
  bf16_t* qb    = (bf16_t*)(ws + 35258368);        // 33554432
  bf16_t* kb    = (bf16_t*)(ws + 68812800);        // 33554432
  bf16_t* vb    = (bf16_t*)(ws + 102367232);       // 33554432
  bf16_t* obuf  = (bf16_t*)(ws + 135921664);       // 33554432
  float*  x1    = (float*) (ws + 169476096);       // 65536*256*4 = 67108864
  bf16_t* h2    = ywin;                            // reuse (ywin dead after QKV)
  bf16_t* h3    = qb;                              // reuse q..obuf (134217728 B)

  // 1) weight conversion + bias table expansion
  cvt_bf16_kernel<<<768, 256, 0, stream>>>(qkv_w, qkvW, 768 * 256);
  cvt_bf16_kernel<<<256, 256, 0, stream>>>(proj_w, projW, 256 * 256);
  cvt_bf16_kernel<<<1024, 256, 0, stream>>>(fc1_w, fc1W, 1024 * 256);
  cvt_bf16_kernel<<<1024, 256, 0, stream>>>(fc2_w, fc2W, 256 * 1024);
  bias_expand_kernel<<<128, 256, 0, stream>>>(relTab, biasE);

  // 2) LN1 + window partition
  ln_kernel<true><<<M / 8, 256, 0, stream>>>(x, n1w, n1b, ywin);

  // 3) QKV GEMM (M=65536, N=768, K=256) with q/k/v scatter
  gemm_bf16_kernel<0><<<dim3(M / 64, 3), 256, 0, stream>>>(
      ywin, qkvW, qkv_b, nullptr, qb, kb, vb, M, 768, 256);

  // 4) windowed attention, 1024 windows x 8 heads
  attn_kernel<<<8192, 256, 0, stream>>>(qb, kb, vb, biasE, obuf);

  // 5) proj GEMM + residual (window-major -> natural), x1 = x + attn_out
  gemm_bf16_kernel<1><<<dim3(M / 64, 1), 256, 0, stream>>>(
      obuf, projW, proj_b, x, x1, nullptr, nullptr, M, 256, 256);

  // 6) LN2
  ln_kernel<false><<<M / 8, 256, 0, stream>>>(x1, n2w, n2b, h2);

  // 7) FC1 + exact GELU (M, N=1024, K=256)
  gemm_bf16_kernel<2><<<dim3(M / 64, 4), 256, 0, stream>>>(
      h2, fc1W, fc1_b, nullptr, h3, nullptr, nullptr, M, 1024, 256);

  // 8) FC2 + residual -> d_out (M, N=256, K=1024)
  gemm_bf16_kernel<3><<<dim3(M / 64, 1), 256, 0, stream>>>(
      h3, fc2W, fc2_b, x1, out, nullptr, nullptr, M, 256, 1024);
}
